// SwinTransformer_17265768529936
// MI455X (gfx1250) — matrix operand order repaired and checked by reading, not compiled, not run on hardware
//
#include <hip/hip_runtime.h>
#include <math.h>

// ---------------------------------------------------------------------------
// CDNA5 (gfx1250) Swin-Transformer forward.
// All matmuls run on V_WMMA_F32_16X16X32_BF16 (wave32, 16x16 tiles, K=32).
// Window attention is fully fused per (window, head) in LDS.
// Staging loads are branch-free float4 (clamped address + zero-select),
// with global_prefetch of the next K-step tiles overlapping the WMMAs.
// ---------------------------------------------------------------------------

typedef __attribute__((ext_vector_type(16))) __bf16 v16bf;
typedef __attribute__((ext_vector_type(8)))  float  v8f;

#define GF_BIAS 1
#define GF_GELU 2
#define GF_RES  4

__device__ __forceinline__ v8f wmma_bf16(v16bf a, v16bf b, v8f c) {
    // 8 args: (neg_a, A, neg_b, B, c_mod, C, reuse_a, reuse_b)
    return __builtin_amdgcn_wmma_f32_16x16x32_bf16(false, a, false, b,
                                                   (short)0, c, false, false);
}

// A-matrix fragment (16x32 bf16, MxK). base points at tile row 0 (+K offset).
// lane l: M = l&15; elements e -> K = (l>>4)*16 + e
__device__ __forceinline__ v16bf frag_a(const __bf16* base, int stride, int lane) {
    int m = lane & 15, half = lane >> 4;
    const __bf16* p0 = base + m * stride + half * 16;
    v16bf a;
#pragma unroll
    for (int e = 0; e < 16; ++e) a[e] = p0[e];
    return a;
}

// B-matrix fragment (32x16 bf16, KxN) read from an N-major LDS tile
// (base[n*stride + k]). lane l: N = l&15; elements e -> K = (l>>4)*16 + e
__device__ __forceinline__ v16bf frag_b(const __bf16* base, int stride, int lane) {
    int n = lane & 15, half = lane >> 4;
    const __bf16* p = base + n * stride + half * 16;
    v16bf b;
#pragma unroll
    for (int e = 0; e < 16; ++e) b[e] = p[e];
    return b;
}

// ---------------------------------------------------------------------------
// Generic GEMM: C[M,N] = act(A[M,K] @ B[K,N] + bias) + residual
// Block tile 128(M) x 32(N), K-step 32. 8 waves: 4(M) x 2(N), each wave owns
// two 16x16 tiles (rows wm*16 and 64+wm*16) sharing one B fragment.
// ---------------------------------------------------------------------------
__global__ __launch_bounds__(256) void gemm_bf16_kernel(
    const float* __restrict__ A, const float* __restrict__ B,
    const float* __restrict__ bias, const float* __restrict__ res,
    float* __restrict__ C, int M, int N, int K, int flags)
{
    __shared__ __bf16 lA[128][40];   // 128 rows x 32 K (+pad)
    __shared__ __bf16 lBt[32][40];   // 32 N    x 32 K (N-major!)

    int tid = threadIdx.x, lane = tid & 31, wid = tid >> 5;
    int wm = wid & 3, wn = wid >> 2;
    int m0 = blockIdx.x * 128, n0 = blockIdx.y * 32;
    v8f acc0 = {}, acc1 = {};

    for (int k0 = 0; k0 < K; k0 += 32) {
        // --- stage 1: branch-free float4 global loads into registers ---
        float4 av[4];
#pragma unroll
        for (int i = 0; i < 4; ++i) {                    // A: 128x32 = 1024 float4
            int idx = tid + i * 256;
            int r = idx >> 3, c4 = (idx & 7) * 4;
            int gm = m0 + r, gk = k0 + c4;
            bool ok = (gm < M) && (gk < K);
            long off = ok ? ((long)gm * K + gk) : 0;
            av[i] = *(const float4*)(A + off);
            if (!ok) av[i] = make_float4(0.f, 0.f, 0.f, 0.f);
        }
        float4 bv;
        {
            int kk = tid >> 3, n4 = (tid & 7) * 4;       // B: 32x32 = 256 float4
            int gk = k0 + kk;
            bool ok = gk < K;
            long off = ok ? ((long)gk * N + n0 + n4) : 0;
            bv = *(const float4*)(B + off);
            if (!ok) bv = make_float4(0.f, 0.f, 0.f, 0.f);
        }
        // --- prefetch next K-step tiles (global_prefetch_b8, no counters) ---
        if (k0 + 32 < K) {
#pragma unroll
            for (int i = 0; i < 4; ++i) {
                int idx = tid + i * 256;
                int r = idx >> 3, c4 = (idx & 7) * 4;
                int gm = m0 + r, gk = k0 + 32 + c4;
                if (gm < M && gk < K)
                    __builtin_prefetch(A + (long)gm * K + gk, 0, 3);
            }
            int kk = tid >> 3, n4 = (tid & 7) * 4;
            int gk = k0 + 32 + kk;
            if (gk < K)
                __builtin_prefetch(B + (long)gk * N + n0 + n4, 0, 3);
        }
        __syncthreads();   // previous iteration's fragment reads finished
        // --- stage 2: convert + store to LDS ---
#pragma unroll
        for (int i = 0; i < 4; ++i) {
            int idx = tid + i * 256;
            int r = idx >> 3, c4 = (idx & 7) * 4;
            lA[r][c4 + 0] = (__bf16)av[i].x;
            lA[r][c4 + 1] = (__bf16)av[i].y;
            lA[r][c4 + 2] = (__bf16)av[i].z;
            lA[r][c4 + 3] = (__bf16)av[i].w;
        }
        {
            int kk = tid >> 3, n4 = (tid & 7) * 4;
            lBt[n4 + 0][kk] = (__bf16)bv.x;
            lBt[n4 + 1][kk] = (__bf16)bv.y;
            lBt[n4 + 2][kk] = (__bf16)bv.z;
            lBt[n4 + 3][kk] = (__bf16)bv.w;
        }
        __syncthreads();
        // --- stage 3: fragments + WMMA ---
        v16bf bf  = frag_b(&lBt[wn * 16][0], 40, lane);
        v16bf af0 = frag_a(&lA[wm * 16][0], 40, lane);
        v16bf af1 = frag_a(&lA[64 + wm * 16][0], 40, lane);
        acc0 = wmma_bf16(af0, bf, acc0);
        acc1 = wmma_bf16(af1, bf, acc1);
    }

    int half = lane >> 4, cn = lane & 15;
    int gn = n0 + wn * 16 + cn;
    float bval = (flags & GF_BIAS) ? bias[gn] : 0.f;
#pragma unroll
    for (int sub = 0; sub < 2; ++sub) {
        v8f acc = sub ? acc1 : acc0;
        int mbase = m0 + sub * 64 + wm * 16;
#pragma unroll
        for (int v = 0; v < 8; ++v) {
            int gm = mbase + v + half * 8;
            if (gm < M) {
                float val = acc[v] + bval;
                if (flags & GF_GELU) val = 0.5f * val * (1.f + erff(val * 0.70710678118654752f));
                if (flags & GF_RES)  val += res[(long)gm * N + gn];
                C[(long)gm * N + gn] = val;
            }
        }
    }
}

// ---------------------------------------------------------------------------
// Fused window attention: one workgroup per (window, head).
// 49x49 scores padded to 64x64 in LDS; shift folded into gather/scatter.
// ---------------------------------------------------------------------------
__global__ __launch_bounds__(256) void swin_attn_kernel(
    const float* __restrict__ qkv, const float* __restrict__ pos,
    float* __restrict__ out, int Hsp, int Wsp, int dim, int heads, int shifted)
{
    __shared__ __bf16 lQ[64][40];    // tokens x head_dim
    __shared__ __bf16 lK[64][40];    // tokens x head_dim (== B fragment layout for Q@K^T)
    __shared__ __bf16 lVt[32][72];   // head_dim x tokens (N-major for P@V)
    __shared__ float  lS[64][64];    // scores
    __shared__ __bf16 lP[64][72];    // softmax(scores), A-matrix for P@V

    int tid = threadIdx.x, lane = tid & 31, wid = tid >> 5;
    int head = blockIdx.y;
    int nwh = Hsp / 7, nww = Wsp / 7, nw = nwh * nww;
    int wi = blockIdx.x;
    int b = wi / nw, wl = wi % nw, wh = wl / nww, ww = wl % nww;
    int stride3 = 3 * dim;

    // Branch-free float4 gather: 64x32 per matrix = 512 float4, 2 per thread.
#pragma unroll
    for (int it = 0; it < 2; ++it) {
        int idx = tid + it * 256;
        int i = idx >> 3, d4 = (idx & 7) * 4;
        int tc = i < 49 ? i : 48;                     // clamp token (stay in bounds)
        int i1 = tc / 7, i2 = tc % 7;
        int y = wh * 7 + i1, x = ww * 7 + i2;
        if (shifted) { y = (y + 3) % Hsp; x = (x + 3) % Wsp; }
        long base = ((long)(b * Hsp + y) * Wsp + x) * stride3 + head * 32 + d4;
        float4 q = *(const float4*)(qkv + base);
        float4 k = *(const float4*)(qkv + base + dim);
        float4 v = *(const float4*)(qkv + base + 2 * dim);
        if (i >= 49) {
            q = make_float4(0.f, 0.f, 0.f, 0.f);
            k = make_float4(0.f, 0.f, 0.f, 0.f);
            v = make_float4(0.f, 0.f, 0.f, 0.f);
        }
        lQ[i][d4 + 0] = (__bf16)q.x; lQ[i][d4 + 1] = (__bf16)q.y;
        lQ[i][d4 + 2] = (__bf16)q.z; lQ[i][d4 + 3] = (__bf16)q.w;
        lK[i][d4 + 0] = (__bf16)k.x; lK[i][d4 + 1] = (__bf16)k.y;
        lK[i][d4 + 2] = (__bf16)k.z; lK[i][d4 + 3] = (__bf16)k.w;
        lVt[d4 + 0][i] = (__bf16)v.x; lVt[d4 + 1][i] = (__bf16)v.y;
        lVt[d4 + 2][i] = (__bf16)v.z; lVt[d4 + 3][i] = (__bf16)v.w;
    }
    __syncthreads();

    // --- phase 1: S = Q @ K^T * scale + rel_pos_bias (+ shift masks) ---
    const float scale = 0.17677669529663687f;       // 1/sqrt(32)
    bool mrow = shifted && (wh == nwh - 1);
    bool mcol = shifted && (ww == nww - 1);
#pragma unroll
    for (int tt = 0; tt < 2; ++tt) {
        int t = wid + tt * 8;                        // 16 tiles over 8 waves
        int mt = t >> 2, nt = t & 3;
        v16bf af = frag_a(&lQ[mt * 16][0], 40, lane);
        v16bf bf = frag_b(&lK[nt * 16][0], 40, lane);
        v8f acc = {};
        acc = wmma_bf16(af, bf, acc);
        int half = lane >> 4, cn = lane & 15;
        int j = nt * 16 + cn;
#pragma unroll
        for (int v = 0; v < 8; ++v) {
            int i = mt * 16 + v + half * 8;
            float s = -1e9f;                         // pad rows/cols -> -inf
            if (i < 49 && j < 49) {
                int i1 = i / 7, i2 = i % 7, j1 = j / 7, j2 = j % 7;
                s = acc[v] * scale + pos[(j1 - i1 + 6) * 13 + (j2 - i2 + 6)];
                if (mrow && ((i >= 28) != (j >= 28))) s += -1e9f;
                if (mcol && ((i2 >= 4) != (j2 >= 4))) s += -1e9f;
            }
            lS[i][j] = s;
        }
    }
    __syncthreads();

    // --- phase 2: row softmax (each wave does 8 rows, 2 cols/lane) ---
#pragma unroll
    for (int rr = 0; rr < 8; ++rr) {
        int row = wid * 8 + rr;
        float a0 = lS[row][lane], a1 = lS[row][lane + 32];
        float mx = fmaxf(a0, a1);
        for (int off = 16; off; off >>= 1) mx = fmaxf(mx, __shfl_xor(mx, off));
        float e0 = expf(a0 - mx), e1 = expf(a1 - mx);
        float s = e0 + e1;
        for (int off = 16; off; off >>= 1) s += __shfl_xor(s, off);
        float inv = 1.f / s;
        lP[row][lane]      = (__bf16)(e0 * inv);
        lP[row][lane + 32] = (__bf16)(e1 * inv);
    }
    __syncthreads();

    // --- phase 3: O = P @ V (8 tiles: 4 M x 2 N, K = 64 in two steps) ---
    {
        int mt = wid & 3, nt = wid >> 2;
        v8f acc = {};
#pragma unroll
        for (int ks = 0; ks < 2; ++ks) {
            v16bf af = frag_a(&lP[mt * 16][ks * 32], 72, lane);
            v16bf bf = frag_b(&lVt[nt * 16][ks * 32], 72, lane);
            acc = wmma_bf16(af, bf, acc);
        }
        int half = lane >> 4, cn = lane & 15;
        int d = nt * 16 + cn;
#pragma unroll
        for (int v = 0; v < 8; ++v) {
            int i = mt * 16 + v + half * 8;
            if (i < 49) {
                int i1 = i / 7, i2 = i % 7;
                int y = wh * 7 + i1, x = ww * 7 + i2;
                if (shifted) { y = (y + 3) % Hsp; x = (x + 3) % Wsp; }
                out[((long)(b * Hsp + y) * Wsp + x) * dim + head * 32 + d] = acc[v];
            }
        }
    }
}

// ---------------------------------------------------------------------------
// LayerNorm: one wave per token.
// ---------------------------------------------------------------------------
__global__ __launch_bounds__(256) void layernorm_kernel(
    const float* __restrict__ x, const float* __restrict__ g,
    const float* __restrict__ beta, float* __restrict__ y, int tokens, int dim)
{
    int wid = threadIdx.x >> 5, lane = threadIdx.x & 31;
    int t = blockIdx.x * 8 + wid;
    if (t >= tokens) return;
    const float* xp = x + (long)t * dim;
    float s = 0.f, ss = 0.f;
    for (int c = lane; c < dim; c += 32) { float v = xp[c]; s += v; ss += v * v; }
    for (int off = 16; off; off >>= 1) { s += __shfl_xor(s, off); ss += __shfl_xor(ss, off); }
    float mean = s / dim;
    float var  = ss / dim - mean * mean;
    float inv  = rsqrtf(var + 1e-5f);
    float* yp = y + (long)t * dim;
    for (int c = lane; c < dim; c += 32)
        yp[c] = (xp[c] - mean) * inv * g[c] + beta[c];
}

// ---------------------------------------------------------------------------
// Patch-merging unfold: gather (b,c,h,w)-strided input into (token, c*df*df).
// ---------------------------------------------------------------------------
__global__ void unfold_kernel(const float* __restrict__ src, float* __restrict__ dst,
                              int nh, int nw, int C, int df, long total,
                              long sb, long sc, long sy, long sx)
{
    long gid = (long)blockIdx.x * blockDim.x + threadIdx.x;
    if (gid >= total) return;
    int KF = C * df * df;
    long t = gid / KF; int kf = (int)(gid % KF);
    int ox = (int)(t % nw); long t2 = t / nw;
    int oy = (int)(t2 % nh); int b = (int)(t2 / nh);
    int dx = kf % df; int k2 = kf / df;
    int dy = k2 % df; int c = k2 / df;
    dst[gid] = src[b * sb + c * sc + (long)(oy * df + dy) * sy + (long)(ox * df + dx) * sx];
}

// ---------------------------------------------------------------------------
// Final (b,h,w,c) -> (b,c,h,w)
// ---------------------------------------------------------------------------
__global__ void nhwc_to_nchw_kernel(const float* __restrict__ in, float* __restrict__ out,
                                    int Bb, int H, int W, int Cc)
{
    long total = (long)Bb * H * W * Cc;
    long gid = (long)blockIdx.x * blockDim.x + threadIdx.x;
    if (gid >= total) return;
    int c = (int)(gid % Cc); long t = gid / Cc;
    int x = (int)(t % W); t /= W;
    int y = (int)(t % H); int b = (int)(t / H);
    out[(((long)b * Cc + c) * H + y) * W + x] = in[gid];
}

// ---------------------------------------------------------------------------
// Host orchestration.
// ---------------------------------------------------------------------------
extern "C" void kernel_launch(void* const* d_in, const int* in_sizes, int n_in,
                              void* d_out, int out_size, void* d_ws, size_t ws_size,
                              hipStream_t stream)
{
    (void)in_sizes; (void)n_in; (void)out_size; (void)ws_size;
    static const int HEADS_[4] = {3, 6, 12, 24};
    static const int LAYERS_[4] = {2, 2, 6, 2};
    static const int DF_[4] = {4, 2, 2, 2};

    int ii = 0;
    const float* x_in = (const float*)d_in[ii++];   // (8,3,224,224)

    float* ws = (float*)d_ws;
    float* buf_unf = ws; ws += 2408448;   // max tokens*KF (stage1: 6272*384)
    float* buf_x   = ws; ws += 2408448;   // max tokens*dim (stage0: 25088*96)
    float* buf_nrm = ws; ws += 2408448;
    float* buf_qkv = ws; ws += 7225344;   // 25088*288
    float* buf_att = ws; ws += 2408448;
    float* buf_mlp = ws; ws += 3211264;   // 25088*128

    const int Bb = 8;
    int inC = 3, Hin = 224, Win = 224, dm = 96;
    const float* src = x_in;
    long sb = (long)3 * 224 * 224, sc = (long)224 * 224, sy = 224, sx = 1;  // NCHW strides

    for (int s = 0; s < 4; ++s) {
        int df = DF_[s], heads = HEADS_[s];
        int Hsp = Hin / df, Wsp = Win / df;
        int tokens = Bb * Hsp * Wsp;
        int KF = inC * df * df;

        const float* pm_w = (const float*)d_in[ii++];
        const float* pm_b = (const float*)d_in[ii++];

        {
            long total = (long)tokens * KF;
            unsigned blocks = (unsigned)((total + 255) / 256);
            unfold_kernel<<<blocks, 256, 0, stream>>>(src, buf_unf, Hsp, Wsp, inC, df,
                                                      total, sb, sc, sy, sx);
        }
        gemm_bf16_kernel<<<dim3((tokens + 127) / 128, dm / 32), 256, 0, stream>>>(
            buf_unf, pm_w, pm_b, nullptr, buf_x, tokens, dm, KF, GF_BIAS);

        for (int j = 0; j < LAYERS_[s]; ++j) {
            const float* ln1_g  = (const float*)d_in[ii++];
            const float* ln1_b  = (const float*)d_in[ii++];
            const float* qkv_w  = (const float*)d_in[ii++];
            const float* pos_e  = (const float*)d_in[ii++];
            const float* out_w  = (const float*)d_in[ii++];
            const float* out_b  = (const float*)d_in[ii++];
            const float* ln2_g  = (const float*)d_in[ii++];
            const float* ln2_b  = (const float*)d_in[ii++];
            const float* mlp_w1 = (const float*)d_in[ii++];
            const float* mlp_b1 = (const float*)d_in[ii++];
            const float* mlp_w2 = (const float*)d_in[ii++];
            const float* mlp_b2 = (const float*)d_in[ii++];
            int shifted = j & 1;

            layernorm_kernel<<<(tokens + 7) / 8, 256, 0, stream>>>(
                buf_x, ln1_g, ln1_b, buf_nrm, tokens, dm);
            gemm_bf16_kernel<<<dim3((tokens + 127) / 128, (3 * dm) / 32), 256, 0, stream>>>(
                buf_nrm, qkv_w, nullptr, nullptr, buf_qkv, tokens, 3 * dm, dm, 0);
            int nwin = (Hsp / 7) * (Wsp / 7);
            swin_attn_kernel<<<dim3(Bb * nwin, heads), 256, 0, stream>>>(
                buf_qkv, pos_e, buf_att, Hsp, Wsp, dm, heads, shifted);
            gemm_bf16_kernel<<<dim3((tokens + 127) / 128, dm / 32), 256, 0, stream>>>(
                buf_att, out_w, out_b, buf_x, buf_x, tokens, dm, dm, GF_BIAS | GF_RES);
            layernorm_kernel<<<(tokens + 7) / 8, 256, 0, stream>>>(
                buf_x, ln2_g, ln2_b, buf_nrm, tokens, dm);
            gemm_bf16_kernel<<<dim3((tokens + 127) / 128, 128 / 32), 256, 0, stream>>>(
                buf_nrm, mlp_w1, mlp_b1, nullptr, buf_mlp, tokens, 128, dm, GF_BIAS | GF_GELU);
            gemm_bf16_kernel<<<dim3((tokens + 127) / 128, dm / 32), 256, 0, stream>>>(
                buf_mlp, mlp_w2, mlp_b2, buf_x, buf_x, tokens, dm, 128, GF_BIAS | GF_RES);
        }

        // next stage reads buf_x as (b, h, w, c)
        src = buf_x;
        inC = dm; Hin = Hsp; Win = Wsp;
        sb = (long)Hsp * Wsp * dm; sc = 1; sy = (long)Wsp * dm; sx = dm;
        dm *= 2;
    }

    nhwc_to_nchw_kernel<<<(8 * 7 * 7 * 768 + 255) / 256, 256, 0, stream>>>(
        buf_x, (float*)d_out, 8, 7, 7, 768);
}
